// LinearAttention_25769803806
// MI455X (gfx1250) — matrix operand order repaired and checked by reading, not compiled, not run on hardware
//
#include <hip/hip_runtime.h>
#include <hip/hip_bf16.h>

typedef _Float16 h16;
typedef __attribute__((ext_vector_type(16))) _Float16 v16h;
typedef __attribute__((ext_vector_type(8)))  _Float16 v8h;
typedef __attribute__((ext_vector_type(8)))  float    v8f;

#define NB_      16
#define NC_      512
#define NN_      4096
#define NHEADS_  8
#define NDH_     64
#define NMEM_    4
#define NGROUPS_ 8
#define CPG_     64
#define GELEMS_  (CPG_ * NN_)     // 262144 elems per (batch, group)
#define QK_SCALE 0.125f           // 64^-0.5
#define BKPAD    40               // 32 K-halves + 8 pad halves (bank spread)

static __device__ __forceinline__ v8f wmma16(v16h a, v16h b, v8f c) {
  return __builtin_amdgcn_wmma_f32_16x16x32_f16(false, a, false, b, (short)0, c,
                                                false, false);
}
static __device__ __forceinline__ v16h cat8(v8h lo, v8h hi) {
  return __builtin_shufflevector(lo, hi, 0, 1, 2, 3, 4, 5, 6, 7, 8, 9, 10, 11,
                                 12, 13, 14, 15);
}

// ---------------------------------------------------------------- f32 -> f16
__global__ __launch_bounds__(256) void cvt_f16_kernel(const float* __restrict__ in,
                                                      h16* __restrict__ out, int n) {
  int i = blockIdx.x * 256 + threadIdx.x;
  if (i < n) out[i] = (h16)in[i];
}

// ------------------------------------------------- GroupNorm1: x -> xn_t[b][n][c] f16
__global__ __launch_bounds__(256) void gn1_kernel(const float* __restrict__ x,
                                                  const float* __restrict__ w,
                                                  const float* __restrict__ bb,
                                                  h16* __restrict__ xn_t) {
  __shared__ float r1[256], r2[256];
  int bg = blockIdx.x;
  int b = bg >> 3, g = bg & 7;
  const float* xp = x + ((size_t)b * NC_ + (size_t)g * CPG_) * NN_;
  int tid = threadIdx.x;
  float s = 0.f, s2 = 0.f;
  for (int i = tid; i < GELEMS_; i += 256) {
    float v = xp[i];
    s += v; s2 += v * v;
  }
  r1[tid] = s; r2[tid] = s2;
  __syncthreads();
  for (int st = 128; st > 0; st >>= 1) {
    if (tid < st) { r1[tid] += r1[tid + st]; r2[tid] += r2[tid + st]; }
    __syncthreads();
  }
  float mean = r1[0] * (1.f / (float)GELEMS_);
  float var  = r2[0] * (1.f / (float)GELEMS_) - mean * mean;
  float rstd = rsqrtf(var + 1e-5f);
  for (int i = tid; i < GELEMS_; i += 256) {
    int c = g * CPG_ + (i >> 12);
    int n = i & (NN_ - 1);
    float v = (xp[i] - mean) * rstd * w[c] + bb[c];
    xn_t[((size_t)b * NN_ + n) * NC_ + c] = (h16)v;
  }
}

// ------------------------------------------------- generic f16 WMMA GEMM
// out[b][m][n] = A[m][k] * Bt[b][n][k] (+bias[m]); block tile 128(M) x 64(N)
// B panel staged through LDS (double-buffered) and shared by the 8 waves.
__global__ __launch_bounds__(256) void gemm_f16_kernel(
    const h16* __restrict__ A, const h16* __restrict__ Bt,
    void* __restrict__ outp, const float* __restrict__ bias,
    int M, int N, int K, int outF16) {
  __shared__ h16 sb[2][64 * BKPAD];

  const int b    = blockIdx.z;
  const int tid  = threadIdx.x;
  const int wave = tid >> 5;
  const int lane = tid & 31;
  const int l    = lane & 15;
  const int half = lane >> 4;
  const int m0   = blockIdx.y * 128 + wave * 16;
  const int n0   = blockIdx.x * 64;

  // staging: thread t copies 8 halves (16B): B row nl = t>>2, quarter q = t&3
  const int nl = tid >> 2, q = tid & 3;
  const h16* gsrc  = Bt + (size_t)b * N * K + (size_t)(n0 + nl) * K + q * 8;
  h16* ldst0 = &sb[0][nl * BKPAD + q * 8];
  h16* ldst1 = &sb[1][nl * BKPAD + q * 8];

  const h16* arow = A + (size_t)(m0 + l) * K + half * 16;
  const int  roff = l * BKPAD + half * 16;  // per-lane B read offset within tile

  v8f acc[4] = {{}, {}, {}, {}};
  if (bias) {
#pragma unroll
    for (int v = 0; v < 8; v++) {
      float bv = bias[m0 + half * 8 + v];
      acc[0][v] = bv; acc[1][v] = bv; acc[2][v] = bv; acc[3][v] = bv;
    }
  }

  // preload K-tile 0 into buffer 0
  {
    v8h st0 = *(const v8h*)(gsrc);
    *(v8h*)ldst0 = st0;
  }
  __syncthreads();

  int cur = 0;
  for (int k0 = 0; k0 < K; k0 += 32) {
    // A loads first: the WMMA-dependent wait can then leave the B prefetch
    // (issued last) outstanding across the compute section.
    v8h alo = *(const v8h*)(arow + k0);
    v8h ahi = *(const v8h*)(arow + k0 + 8);
    // branch-free prefetch: last iteration harmlessly re-reads tile 0
    const int kp = (k0 + 32 < K) ? (k0 + 32) : 0;
    v8h nxt = *(const v8h*)(gsrc + kp);
    v16h a  = cat8(alo, ahi);

    const h16* base = &sb[cur][0];
#pragma unroll
    for (int nt = 0; nt < 4; nt++) {
      const h16* p = base + nt * 16 * BKPAD + roff;
      v16h bmat = cat8(*(const v8h*)p, *(const v8h*)(p + 8));
      acc[nt] = wmma16(a, bmat, acc[nt]);
    }
    __syncthreads();                          // all waves done reading buf[cur]
    *(v8h*)(cur ? ldst0 : ldst1) = nxt;       // unconditional store to buf[nxt]
    __syncthreads();                          // staged writes visible
    cur ^= 1;
  }

  size_t ob = (size_t)b * M * N;
  if (outF16) {
    h16* o = (h16*)outp + ob;
#pragma unroll
    for (int v = 0; v < 8; v++) {
      size_t m = (size_t)(m0 + half * 8 + v) * N + n0 + l;
      o[m]      = (h16)acc[0][v];
      o[m + 16] = (h16)acc[1][v];
      o[m + 32] = (h16)acc[2][v];
      o[m + 48] = (h16)acc[3][v];
    }
  } else {
    float* o = (float*)outp + ob;
#pragma unroll
    for (int v = 0; v < 8; v++) {
      size_t m = (size_t)(m0 + half * 8 + v) * N + n0 + l;
      o[m]      = acc[0][v];
      o[m + 16] = acc[1][v];
      o[m + 32] = acc[2][v];
      o[m + 48] = acc[3][v];
    }
  }
}

// ------------------------------------------------- k softmax stats (in-place exp)
// one block per row (b,h,d); row length NN_ (+ NMEM_ memory slots)
__global__ __launch_bounds__(256) void kstat_kernel(h16* __restrict__ qkv_h,
                                                    const float* __restrict__ mem_kv,
                                                    float* __restrict__ memk_exp,
                                                    float* __restrict__ rksum) {
  __shared__ float red[256];
  int row = blockIdx.x;            // b*512 + h*64 + d
  int b = row >> 9, hd = row & 511;
  h16* rk = qkv_h + ((size_t)b * (3 * NC_) + NC_ + hd) * NN_;
  const float* mk = mem_kv + (size_t)hd * NMEM_;  // mem_kv[0][h][d][:]
  int tid = threadIdx.x;

  float mx = -1e30f;
  for (int i = tid; i < NN_; i += 256) mx = fmaxf(mx, (float)rk[i]);
  if (tid < NMEM_) mx = fmaxf(mx, mk[tid]);
  red[tid] = mx;
  __syncthreads();
  for (int st = 128; st > 0; st >>= 1) {
    if (tid < st) red[tid] = fmaxf(red[tid], red[tid + st]);
    __syncthreads();
  }
  float gmax = red[0];
  __syncthreads();

  float s = 0.f;
  for (int i = tid; i < NN_; i += 256) {
    float e = __expf((float)rk[i] - gmax);
    rk[i] = (h16)e;
    s += e;
  }
  if (tid < NMEM_) {
    float e = __expf(mk[tid] - gmax);
    memk_exp[(size_t)row * NMEM_ + tid] = e;
    s += e;
  }
  red[tid] = s;
  __syncthreads();
  for (int st = 128; st > 0; st >>= 1) {
    if (tid < st) red[tid] += red[tid + st];
    __syncthreads();
  }
  if (tid == 0) rksum[row] = 1.f / red[0];
}

// ------------------------------------------------- q softmax over d, write qsmT[b][h][n][d]
__global__ __launch_bounds__(256) void qsm_kernel(const h16* __restrict__ qkv_h,
                                                  h16* __restrict__ qsmT) {
  int idx = blockIdx.x * 256 + threadIdx.x;   // over NB_*NHEADS_*NN_
  int n = idx & (NN_ - 1);
  int bh = idx >> 12;
  int b = bh >> 3, h = bh & 7;
  const h16* qp = qkv_h + ((size_t)b * (3 * NC_) + (size_t)h * NDH_) * NN_ + n;
  float mx = -1e30f;
  for (int d = 0; d < NDH_; d++) mx = fmaxf(mx, (float)qp[(size_t)d * NN_]);
  float s = 0.f;
  for (int d = 0; d < NDH_; d++) s += __expf((float)qp[(size_t)d * NN_] - mx);
  float rs = QK_SCALE / s;
  h16* op = qsmT + ((size_t)bh * NN_ + n) * NDH_;
  for (int d = 0; d < NDH_; d++)
    op[d] = (h16)(__expf((float)qp[(size_t)d * NN_] - mx) * rs);
}

// ------------------------------------------------- context: ctxT[b][h][e][d] f16
__global__ __launch_bounds__(256) void ctx_kernel(const h16* __restrict__ qkv_h,
                                                  const float* __restrict__ mem_kv,
                                                  const float* __restrict__ memk_exp,
                                                  const float* __restrict__ rksum,
                                                  h16* __restrict__ ctxT) {
  int bh = blockIdx.x;
  int b = bh >> 3, h = bh & 7;
  int tid = threadIdx.x, wave = tid >> 5, lane = tid & 31;
  int l = lane & 15, half = lane >> 4;
  int m0 = (wave & 3) * 16;        // d tile
  int e0 = (wave >> 2) * 32;       // e tiles: e0, e0+16

  const h16* kbase = qkv_h + ((size_t)b * (3 * NC_) + NC_ + (size_t)h * NDH_) * NN_;
  const h16* vbase = qkv_h + ((size_t)b * (3 * NC_) + 2 * NC_ + (size_t)h * NDH_) * NN_;
  const h16* arow = kbase + (size_t)(m0 + l) * NN_ + half * 16;
  const h16* b0r  = vbase + (size_t)(e0 + l) * NN_ + half * 16;
  const h16* b1r  = b0r + (size_t)16 * NN_;

  v8f acc[2] = {{}, {}};
  for (int k0 = 0; k0 < NN_; k0 += 32) {
    v8h alo = *(const v8h*)(arow + k0);
    v8h ahi = *(const v8h*)(arow + k0 + 8);
    v16h a  = cat8(alo, ahi);
    v16h bb0 = *(const v16h*)(b0r + k0);
    v16h bb1 = *(const v16h*)(b1r + k0);
    acc[0] = wmma16(a, bb0, acc[0]);
    acc[1] = wmma16(a, bb1, acc[1]);
  }
#pragma unroll
  for (int et = 0; et < 2; et++) {
    int e = e0 + et * 16 + l;
    const float* mv = mem_kv + ((size_t)(NHEADS_ + h) * NDH_ + e) * NMEM_;  // mem_kv[1][h][e][:]
    float mv0 = mv[0], mv1 = mv[1], mv2 = mv[2], mv3 = mv[3];
    v8h ov;
#pragma unroll
    for (int v = 0; v < 8; v++) {
      int d = m0 + half * 8 + v;
      const float* me = memk_exp + ((size_t)bh * NDH_ + d) * NMEM_;
      float extra = me[0] * mv0 + me[1] * mv1 + me[2] * mv2 + me[3] * mv3;
      float val = (acc[et][v] + extra) * rksum[(size_t)bh * NDH_ + d];
      ov[v] = (h16)val;
    }
    *(v8h*)(ctxT + ((size_t)bh * NDH_ + e) * NDH_ + m0 + half * 8) = ov;
  }
}

// ------------------------------------------------- attn: attn_t[b][n][h*64+e] f16
__global__ __launch_bounds__(256) void attn_kernel(const h16* __restrict__ ctxT,
                                                   const h16* __restrict__ qsmT,
                                                   h16* __restrict__ attn_t) {
  int bh = blockIdx.y;
  int b = bh >> 3, h = bh & 7;
  int tid = threadIdx.x, wave = tid >> 5, lane = tid & 31;
  int l = lane & 15, half = lane >> 4;
  int m0 = (wave & 3) * 16;                       // e tile
  int n0 = blockIdx.x * 32 + (wave >> 2) * 16;    // n tile

  const h16* arow = ctxT + ((size_t)bh * NDH_ + m0 + l) * NDH_ + half * 16;
  const h16* brow = qsmT + ((size_t)bh * NN_ + n0 + l) * NDH_ + half * 16;
  v8f acc = {};
#pragma unroll
  for (int k0 = 0; k0 < NDH_; k0 += 32) {
    v8h alo = *(const v8h*)(arow + k0);
    v8h ahi = *(const v8h*)(arow + k0 + 8);
    v16h a  = cat8(alo, ahi);
    v16h bb = *(const v16h*)(brow + k0);
    acc = wmma16(a, bb, acc);
  }
  v8h ov;
#pragma unroll
  for (int v = 0; v < 8; v++) ov[v] = (h16)acc[v];
  *(v8h*)(attn_t + ((size_t)b * NN_ + n0 + l) * NC_ + h * NDH_ + m0 + half * 8) = ov;
}

// ------------------------------------------------- GroupNorm2: out_proj f32 -> d_out f32
__global__ __launch_bounds__(256) void gn2_kernel(const float* __restrict__ y,
                                                  const float* __restrict__ w,
                                                  const float* __restrict__ bb,
                                                  float* __restrict__ outp) {
  __shared__ float r1[256], r2[256];
  int bg = blockIdx.x;
  int b = bg >> 3, g = bg & 7;
  size_t base = ((size_t)b * NC_ + (size_t)g * CPG_) * NN_;
  const float* yp = y + base;
  float* op = outp + base;
  int tid = threadIdx.x;
  float s = 0.f, s2 = 0.f;
  for (int i = tid; i < GELEMS_; i += 256) {
    float v = yp[i];
    s += v; s2 += v * v;
  }
  r1[tid] = s; r2[tid] = s2;
  __syncthreads();
  for (int st = 128; st > 0; st >>= 1) {
    if (tid < st) { r1[tid] += r1[tid + st]; r2[tid] += r2[tid + st]; }
    __syncthreads();
  }
  float mean = r1[0] * (1.f / (float)GELEMS_);
  float var  = r2[0] * (1.f / (float)GELEMS_) - mean * mean;
  float rstd = rsqrtf(var + 1e-5f);
  for (int i = tid; i < GELEMS_; i += 256) {
    int c = g * CPG_ + (i >> 12);
    op[i] = (yp[i] - mean) * rstd * w[c] + bb[c];
  }
}

// ----------------------------------------------------------------------------
extern "C" void kernel_launch(void* const* d_in, const int* in_sizes, int n_in,
                              void* d_out, int out_size, void* d_ws, size_t ws_size,
                              hipStream_t stream) {
  const float* x      = (const float*)d_in[0];
  const float* gn1_w  = (const float*)d_in[1];
  const float* gn1_b  = (const float*)d_in[2];
  const float* mem_kv = (const float*)d_in[3];
  const float* w_qkv  = (const float*)d_in[4];
  const float* w_out  = (const float*)d_in[5];
  const float* b_out  = (const float*)d_in[6];
  const float* gn2_w  = (const float*)d_in[7];
  const float* gn2_b  = (const float*)d_in[8];
  float* out = (float*)d_out;

  char* ws = (char*)d_ws;
  // region A: xn_t (f16, 64MB) overlaid later by out_proj (f32, 128MB)
  const size_t SZ_REGA  = (size_t)NB_ * NC_ * NN_ * 4;
  const size_t OFF_QKV  = SZ_REGA;
  const size_t SZ_QKV   = (size_t)NB_ * 3 * NC_ * NN_ * 2;
  const size_t OFF_QSMT = OFF_QKV + SZ_QKV;
  const size_t SZ_QSMT  = (size_t)NB_ * NHEADS_ * NN_ * NDH_ * 2;
  const size_t OFF_ATT  = OFF_QSMT + SZ_QSMT;
  const size_t SZ_ATT   = (size_t)NB_ * NN_ * NC_ * 2;
  const size_t OFF_CTX  = OFF_ATT + SZ_ATT;
  const size_t SZ_CTX   = (size_t)NB_ * NHEADS_ * NDH_ * NDH_ * 2;
  const size_t OFF_MEMK = OFF_CTX + SZ_CTX;
  const size_t SZ_MEMK  = (size_t)NB_ * NHEADS_ * NDH_ * NMEM_ * 4;
  const size_t OFF_RKS  = OFF_MEMK + SZ_MEMK;
  const size_t SZ_RKS   = (size_t)NB_ * NHEADS_ * NDH_ * 4;
  const size_t OFF_WQKV = OFF_RKS + SZ_RKS;
  const size_t SZ_WQKV  = (size_t)3 * NC_ * NC_ * 2;
  const size_t OFF_WOUT = OFF_WQKV + SZ_WQKV;

  h16*   xn_t     = (h16*)(ws + 0);
  float* out_proj = (float*)(ws + 0);
  h16*   qkv_h    = (h16*)(ws + OFF_QKV);
  h16*   qsmT     = (h16*)(ws + OFF_QSMT);
  h16*   attn_t   = (h16*)(ws + OFF_ATT);
  h16*   ctxT     = (h16*)(ws + OFF_CTX);
  float* memk_exp = (float*)(ws + OFF_MEMK);
  float* rksum    = (float*)(ws + OFF_RKS);
  h16*   w_qkv_h  = (h16*)(ws + OFF_WQKV);
  h16*   w_out_h  = (h16*)(ws + OFF_WOUT);

  // 1) weight conversion
  cvt_f16_kernel<<<(3 * NC_ * NC_ + 255) / 256, 256, 0, stream>>>(w_qkv, w_qkv_h, 3 * NC_ * NC_);
  cvt_f16_kernel<<<(NC_ * NC_ + 255) / 256, 256, 0, stream>>>(w_out, w_out_h, NC_ * NC_);
  // 2) GroupNorm1 -> xn_t
  gn1_kernel<<<NB_ * NGROUPS_, 256, 0, stream>>>(x, gn1_w, gn1_b, xn_t);
  // 3) QKV GEMM: [1536x512] x [512x4096] per batch -> qkv_h f16
  gemm_f16_kernel<<<dim3(NN_ / 64, (3 * NC_) / 128, NB_), 256, 0, stream>>>(
      w_qkv_h, xn_t, (void*)qkv_h, nullptr, 3 * NC_, NN_, NC_, 1);
  // 4) k softmax stats (in-place exp) + memory-slot exp + 1/sum
  kstat_kernel<<<NB_ * NHEADS_ * NDH_, 256, 0, stream>>>(qkv_h, mem_kv, memk_exp, rksum);
  // 5) q softmax over d -> qsmT (SCALE folded in)
  qsm_kernel<<<(NB_ * NHEADS_ * NN_) / 256, 256, 0, stream>>>(qkv_h, qsmT);
  // 6) context = kexp x v^T (+ memory slots, row scale) -> ctxT
  ctx_kernel<<<NB_ * NHEADS_, 256, 0, stream>>>(qkv_h, mem_kv, memk_exp, rksum, ctxT);
  // 7) out = ctx^T x q -> attn_t[b][n][hidden]
  attn_kernel<<<dim3(NN_ / 32, NB_ * NHEADS_), 256, 0, stream>>>(ctxT, qsmT, attn_t);
  // 8) out projection + bias -> out_proj f32 (overlays xn_t region)
  gemm_f16_kernel<<<dim3(NN_ / 64, NC_ / 128, NB_), 256, 0, stream>>>(
      w_out_h, attn_t, (void*)out_proj, b_out, NC_, NN_, NC_, 0);
  // 9) GroupNorm2 -> d_out
  gn2_kernel<<<NB_ * NGROUPS_, 256, 0, stream>>>(out_proj, gn2_w, gn2_b, out);
}
